// TransformerBlock_63333587747063
// MI455X (gfx1250) — compile-verified, matrix operand order checked by reading, not run on hardware
//
#include <hip/hip_runtime.h>
#include <hip/hip_bf16.h>
#include <math.h>

// ---------------------------------------------------------------------------
// Problem constants (B,S,D,H fixed by the reference)
// ---------------------------------------------------------------------------
constexpr int B_  = 4;
constexpr int S_  = 2048;
constexpr int D_  = 768;
constexpr int H_  = 12;
constexpr int DH_ = 64;        // D_/H_
constexpr int FF_ = 4 * D_;    // 3072
constexpr int ROWS_ = B_ * S_; // 8192

typedef __attribute__((ext_vector_type(16))) __bf16 v16bf;
typedef __attribute__((ext_vector_type(8)))  float  v8f;
typedef __attribute__((ext_vector_type(4)))  unsigned int v4u;
typedef __attribute__((ext_vector_type(8)))  int    v8i;
typedef __attribute__((ext_vector_type(4)))  int    v4i;

union Frag16 { v16bf v; uint4 q[2]; };

// Load a 16x32 A-fragment (or 32x16 B-fragment from [n][k] storage).
// ISA layout: lanes 0-15 -> row = lane, K in {0..7,16..23};
//             lanes 16-31 -> row = lane-16, K in {8..15,24..31}.
__device__ __forceinline__ v16bf load_frag(const __bf16* base, int ld, int lane) {
  const int r  = lane & 15;
  const int hf = (lane >> 4) & 1;
  const __bf16* p = base + r * ld + hf * 8;
  Frag16 f;
  f.q[0] = *reinterpret_cast<const uint4*>(p);
  f.q[1] = *reinterpret_cast<const uint4*>(p + 16);
  return f.v;
}

__device__ __forceinline__ v8f wmma_bf16(v16bf a, v16bf b, v8f c) {
  return __builtin_amdgcn_wmma_f32_16x16x32_bf16(false, a, false, b,
                                                 (short)0, c, false, false);
}

// ---------------------------------------------------------------------------
// Tensor Data Mover staging (gfx1250). 2-D tile load, 2-byte elements.
// D# packed per CDNA5 ISA §8: group0 = {count/lds_addr/global_addr/type},
// group1 = {data_size, tensor_dim0/1, tile_dim0/1, dim0_stride}.
// This toolchain exposes the 6-arg builtin:
//   (uint32x4 g0, int32x8 g1, int32x4 g2, int32x4 g3, int32x8 g4, i32 cpol)
// ---------------------------------------------------------------------------
#if defined(__gfx1250__) && __has_builtin(__builtin_amdgcn_tensor_load_to_lds) && \
    __has_builtin(__builtin_amdgcn_s_wait_tensorcnt)
#define USE_TDM 1
#else
#define USE_TDM 0
#endif

#if USE_TDM
__device__ __forceinline__ void tdm_load_2d(unsigned lds_byte_off, const void* gaddr,
                                            unsigned tensor_d0, unsigned tensor_d1,
                                            unsigned tile_d0, unsigned tile_d1,
                                            unsigned long long stride0_elems) {
  const unsigned long long ga = (unsigned long long)(uintptr_t)gaddr;
  v4u g0;
  g0[0] = 1u;                                  // count=1, user descriptor
  g0[1] = lds_byte_off;                        // lds_addr
  g0[2] = (unsigned)ga;                        // global_addr[31:0]
  g0[3] = (unsigned)((ga >> 32) & 0x01FFFFFFu) // global_addr[56:32]
          | 0x80000000u;                       // type=2 ("image")
  v8i g1;
  g1[0] = (int)(1u << 16);                     // workgroup_mask=0, data_size=1 (2B)
  g1[1] = (int)((tensor_d0 & 0xFFFFu) << 16);  // tensor_dim0[15:0] @ bits63:48
  g1[2] = (int)((tensor_d0 >> 16) | ((tensor_d1 & 0xFFFFu) << 16));
  g1[3] = (int)((tensor_d1 >> 16) | (tile_d0 << 16));        // tile_dim0 @127:112
  g1[4] = (int)(tile_d1 & 0xFFFFu);            // tile_dim1; tile_dim2=0
  g1[5] = (int)(unsigned)stride0_elems;        // dim0_stride[31:0]
  g1[6] = (int)(unsigned)((stride0_elems >> 32) & 0xFFFFu);  // stride[47:32]
  g1[7] = 0;
  v4i z4 = {0, 0, 0, 0};
  v8i z8 = {0, 0, 0, 0, 0, 0, 0, 0};
  __builtin_amdgcn_tensor_load_to_lds(g0, g1, z4, z4, z8, 0);
}
__device__ __forceinline__ unsigned lds_off_of(const void* p) {
  return (unsigned)(uintptr_t)p;               // LDS generic addr low bits
}
#endif

// ---------------------------------------------------------------------------
// f32 -> bf16 conversions
// ---------------------------------------------------------------------------
__global__ void cvt_bf16_kernel(const float* __restrict__ in,
                                __bf16* __restrict__ out, int n) {
  int i = blockIdx.x * blockDim.x + threadIdx.x;
  if (i < n) out[i] = (__bf16)in[i];
}

// weights come in as [K][N]; store bf16 transposed [N][K] so GEMM B-tiles are
// contiguous-K copies (no LDS element transpose in the hot loop).
__global__ void cvt_bf16_t_kernel(const float* __restrict__ in,
                                  __bf16* __restrict__ out, int K, int N) {
  int i = blockIdx.x * blockDim.x + threadIdx.x;
  if (i < K * N) {
    int k = i / N, n = i % N;
    out[(size_t)n * K + k] = (__bf16)in[i];
  }
}

// ---------------------------------------------------------------------------
// Row LayerNorm (D_=768): 256 threads/row, 3 elems/thread, bf16 output.
// ---------------------------------------------------------------------------
__global__ __launch_bounds__(256)
void ln_bf16_kernel(const float* __restrict__ x, const float* __restrict__ g,
                    const float* __restrict__ bta, __bf16* __restrict__ out) {
  const int tid = threadIdx.x;
  const int row = blockIdx.x;
  const float* xr = x + (size_t)row * D_;
  float v0 = xr[tid], v1 = xr[tid + 256], v2 = xr[tid + 512];

  __shared__ float red[8];
  __shared__ float stat[2];

  float s = v0 + v1 + v2;
#pragma unroll
  for (int m = 16; m; m >>= 1) s += __shfl_xor(s, m);
  if ((tid & 31) == 0) red[tid >> 5] = s;
  __syncthreads();
  if (tid == 0) {
    float t = 0.f;
#pragma unroll
    for (int i = 0; i < 8; ++i) t += red[i];
    stat[0] = t * (1.0f / D_);
  }
  __syncthreads();
  const float mu = stat[0];
  const float d0 = v0 - mu, d1 = v1 - mu, d2 = v2 - mu;
  s = d0 * d0 + d1 * d1 + d2 * d2;
#pragma unroll
  for (int m = 16; m; m >>= 1) s += __shfl_xor(s, m);
  if ((tid & 31) == 0) red[tid >> 5] = s;
  __syncthreads();
  if (tid == 0) {
    float t = 0.f;
#pragma unroll
    for (int i = 0; i < 8; ++i) t += red[i];
    stat[1] = t * (1.0f / D_);
  }
  __syncthreads();
  const float rstd = rsqrtf(stat[1] + 1e-5f);
  __bf16* o = out + (size_t)row * D_;
  o[tid]       = (__bf16)(d0 * rstd * g[tid]       + bta[tid]);
  o[tid + 256] = (__bf16)(d1 * rstd * g[tid + 256] + bta[tid + 256]);
  o[tid + 512] = (__bf16)(d2 * rstd * g[tid + 512] + bta[tid + 512]);
}

// ---------------------------------------------------------------------------
// WMMA GEMM: C[M,N] = A[M,K](bf16) @ Bt[N,K](bf16)^T + bias, with epilogues.
// Block = 256 threads = 8 wave32s (4x2); tile 128(M) x 64(N) x 32(K).
// Each wave owns a 32x32 strip: 2 A-frags x 2 B-frags -> 4 WMMAs / K-step.
// Tiles staged via a double-buffered Tensor Data Mover pipeline: the next
// K-step's DMA is in flight (s_wait_tensorcnt 2) while this one computes.
// ---------------------------------------------------------------------------
enum { EPI_QKV = 0, EPI_QKV_T = 1, EPI_RESID = 2, EPI_GELU = 3 };

template <int MODE>
__global__ __launch_bounds__(256)
void gemm_wmma_kernel(const __bf16* __restrict__ A, const __bf16* __restrict__ Bt,
                      const float* __restrict__ bias, const float* __restrict__ res,
                      void* __restrict__ outv, int M, int N, int K) {
  __shared__ alignas(16) __bf16 a_lds[2][128 * 32];  // [m][k]
  __shared__ alignas(16) __bf16 b_lds[2][64 * 32];   // [n][k]

  const int tid  = threadIdx.x;
  const int lane = tid & 31;
  const int wave = tid >> 5;
  const int wm   = wave & 3;   // 0..3 : 32-row strip
  const int wn   = wave >> 2;  // 0..1 : 32-col strip
  const int rowBlock = blockIdx.y * 128;
  const int colBlock = blockIdx.x * 64;

  v8f c00 = {}, c01 = {}, c10 = {}, c11 = {};
  const int nk = K >> 5;

#if USE_TDM
  const unsigned a_off[2] = {lds_off_of(&a_lds[0][0]), lds_off_of(&a_lds[1][0])};
  const unsigned b_off[2] = {lds_off_of(&b_lds[0][0]), lds_off_of(&b_lds[1][0])};
  if (wave == 0) {  // prime the pipeline: stage 0 into buffer 0
    tdm_load_2d(a_off[0], &A[(size_t)rowBlock * K], K, M, 32u, 128u, K);
    tdm_load_2d(b_off[0], &Bt[(size_t)colBlock * K], K, N, 32u, 64u, K);
  }
#else
  const int ar = tid >> 2, ac = (tid & 3) * 8;
#endif

  for (int ki = 0; ki < nk; ++ki) {
    const int cur = ki & 1;
#if USE_TDM
    if (wave == 0) {
      if (ki + 1 < nk) {  // issue next stage into the other buffer
        const int k0n = (ki + 1) * 32;
        tdm_load_2d(a_off[cur ^ 1], &A[(size_t)rowBlock * K + k0n], K, M, 32u, 128u, K);
        tdm_load_2d(b_off[cur ^ 1], &Bt[(size_t)colBlock * K + k0n], K, N, 32u, 64u, K);
        __builtin_amdgcn_s_wait_tensorcnt(2);  // current pair landed, next in flight
      } else {
        __builtin_amdgcn_s_wait_tensorcnt(0);
      }
    }
    __syncthreads();  // release compute waves; data for `cur` visible
#else
    const int k0 = ki * 32;
    __syncthreads();  // previous K-step fully consumed before overwrite
    *reinterpret_cast<uint4*>(&a_lds[cur][ar * 32 + ac]) =
        *reinterpret_cast<const uint4*>(&A[(size_t)(rowBlock + ar) * K + k0 + ac]);
    *reinterpret_cast<uint4*>(&a_lds[cur][(ar + 64) * 32 + ac]) =
        *reinterpret_cast<const uint4*>(&A[(size_t)(rowBlock + ar + 64) * K + k0 + ac]);
    *reinterpret_cast<uint4*>(&b_lds[cur][ar * 32 + ac]) =
        *reinterpret_cast<const uint4*>(&Bt[(size_t)(colBlock + ar) * K + k0 + ac]);
    if (k0 + 32 < K) {
      __builtin_prefetch(&A[(size_t)(rowBlock + ar) * K + k0 + 32 + ac], 0, 0);
      __builtin_prefetch(&Bt[(size_t)(colBlock + ar) * K + k0 + 32 + ac], 0, 0);
    }
    __syncthreads();
#endif

    const v16bf a0 = load_frag(&a_lds[cur][(wm * 32) * 32], 32, lane);
    const v16bf a1 = load_frag(&a_lds[cur][(wm * 32 + 16) * 32], 32, lane);
    const v16bf b0 = load_frag(&b_lds[cur][(wn * 32) * 32], 32, lane);
    const v16bf b1 = load_frag(&b_lds[cur][(wn * 32 + 16) * 32], 32, lane);
    c00 = wmma_bf16(a0, b0, c00);
    c01 = wmma_bf16(a0, b1, c01);
    c10 = wmma_bf16(a1, b0, c10);
    c11 = wmma_bf16(a1, b1, c11);
#if USE_TDM
    __syncthreads();  // all waves done reading `cur` => safe DMA target next iter
#endif
  }

  // Epilogue. C layout: slot i -> row = i + 8*(lane>=16), col = lane&15.
  const int hf  = lane >> 4;
  const int col = lane & 15;

#pragma unroll
  for (int r = 0; r < 2; ++r) {
#pragma unroll
    for (int t = 0; t < 2; ++t) {
      v8f acc = (r == 0) ? (t == 0 ? c00 : c01) : (t == 0 ? c10 : c11);
      const int gc = colBlock + wn * 32 + t * 16 + col;
      const float bv = bias[gc];
      const int gRow0 = rowBlock + wm * 32 + r * 16;
#pragma unroll
      for (int i = 0; i < 8; ++i) {
        const int gr = gRow0 + i + hf * 8;
        float v = acc[i] + bv;
        const size_t idx = (size_t)gr * N + gc;
        if constexpr (MODE == EPI_QKV) {
          const int b = gr / S_, s = gr % S_;
          const int h = gc / DH_, d = gc % DH_;
          ((__bf16*)outv)[(((size_t)(b * H_ + h) * S_) + s) * DH_ + d] = (__bf16)v;
        } else if constexpr (MODE == EPI_QKV_T) {  // V stored [B,H,DH,S]
          const int b = gr / S_, s = gr % S_;
          const int h = gc / DH_, d = gc % DH_;
          ((__bf16*)outv)[(((size_t)(b * H_ + h) * DH_) + d) * S_ + s] = (__bf16)v;
        } else if constexpr (MODE == EPI_RESID) {
          ((float*)outv)[idx] = v + res[idx];
        } else {  // EPI_GELU, exact
          const float ge = 0.5f * v * (1.0f + erff(v * 0.70710678118654752f));
          ((__bf16*)outv)[idx] = (__bf16)ge;
        }
      }
    }
  }
}

// ---------------------------------------------------------------------------
// Causal flash attention. Q,K are [B,H,S,DH] bf16; V is [B,H,DH,S] bf16.
// Output [B,S,D] bf16. Block = 256 threads (8 waves): 128 query rows,
// 32-key chunks streamed through LDS (TDM-staged) with online softmax.
// ---------------------------------------------------------------------------
__global__ __launch_bounds__(256)
void attn_kernel(const __bf16* __restrict__ Q, const __bf16* __restrict__ Kx,
                 const __bf16* __restrict__ Vt, __bf16* __restrict__ O) {
  __shared__ alignas(16) __bf16 k_lds[32 * 64];     // [key][dh]
  __shared__ alignas(16) __bf16 vT_lds[64 * 32];    // [dh][key]
  __shared__ alignas(16) __bf16 p_lds[8 * 16 * 32]; // per-wave P tile [16][32]

  const int tid  = threadIdx.x;
  const int lane = tid & 31;
  const int wave = tid >> 5;
  const int hf   = lane >> 4;
  const int col  = lane & 15;

  const int bh    = blockIdx.x;  // b*H + h
  const int qtile = blockIdx.y;  // 128-row query tile
  const int b = bh / H_, h = bh % H_;

  const __bf16* Qp = Q + (size_t)bh * S_ * DH_;
  const __bf16* Kp = Kx + (size_t)bh * S_ * DH_;
  const __bf16* Vp = Vt + (size_t)bh * DH_ * S_;

  const int qbase = qtile * 128 + wave * 16;

  const v16bf qa0 = load_frag(Qp + (size_t)qbase * DH_, DH_, lane);      // dh 0..31
  const v16bf qa1 = load_frag(Qp + (size_t)qbase * DH_ + 32, DH_, lane); // dh 32..63

  float mrow[8], lrow[8];
  v8f o0 = {}, o1 = {}, o2 = {}, o3 = {};
#pragma unroll
  for (int i = 0; i < 8; ++i) { mrow[i] = -3.0e38f; lrow[i] = 0.f; }

#if !USE_TDM
  const int kr = tid >> 3, kcc = (tid & 7) * 8;  // K chunk: 32 rows x 64
  const int vr = tid >> 2, vcc = (tid & 3) * 8;  // V chunk: 64 rows x 32
#endif
  const int nchunks = qtile * 4 + 4;

  __bf16* pw = p_lds + wave * 16 * 32;

  for (int c = 0; c < nchunks; ++c) {
    const int kc = c * 32;
    __syncthreads();
#if USE_TDM
    if (wave == 0) {
      // K chunk: 32 rows of [S][DH] starting at key=kc -> [32][64] in LDS
      tdm_load_2d(lds_off_of(k_lds), &Kp[(size_t)kc * DH_],
                  (unsigned)DH_, (unsigned)S_, 64u, 32u, (unsigned long long)DH_);
      // V chunk: cols kc..kc+31 of [DH][S] -> [64][32] in LDS
      tdm_load_2d(lds_off_of(vT_lds), &Vp[kc],
                  (unsigned)S_, (unsigned)DH_, 32u, 64u, (unsigned long long)S_);
      __builtin_amdgcn_s_wait_tensorcnt(0);
    }
#else
    *reinterpret_cast<uint4*>(&k_lds[kr * 64 + kcc]) =
        *reinterpret_cast<const uint4*>(&Kp[(size_t)(kc + kr) * DH_ + kcc]);
    *reinterpret_cast<uint4*>(&vT_lds[vr * 32 + vcc]) =
        *reinterpret_cast<const uint4*>(&Vp[(size_t)vr * S_ + kc + vcc]);
#endif
    __syncthreads();

    // scores: S[16 q][32 keys] = Q(16xDH) . K^T
    v8f s0 = {}, s1 = {};
    s0 = wmma_bf16(qa0, load_frag(k_lds, 64, lane), s0);
    s0 = wmma_bf16(qa1, load_frag(k_lds + 32, 64, lane), s0);
    s1 = wmma_bf16(qa0, load_frag(k_lds + 16 * 64, 64, lane), s1);
    s1 = wmma_bf16(qa1, load_frag(k_lds + 16 * 64 + 32, 64, lane), s1);

    // online softmax (rows striped: slot i -> row i + 8*hf)
    const float scale = 0.125f;  // 1/sqrt(64)
#pragma unroll
    for (int i = 0; i < 8; ++i) {
      const int r = qbase + i + hf * 8;
      float v0 = s0[i] * scale;
      float v1 = s1[i] * scale;
      if (kc + col > r)      v0 = -3.0e38f;
      if (kc + 16 + col > r) v1 = -3.0e38f;
      float cm = fmaxf(v0, v1);
#pragma unroll
      for (int m = 8; m; m >>= 1) cm = fmaxf(cm, __shfl_xor(cm, m));  // 16-lane
      const float mn = fmaxf(mrow[i], cm);
      const float al = __expf(mrow[i] - mn);
      const float p0 = __expf(v0 - mn);
      const float p1 = __expf(v1 - mn);
      float rs = p0 + p1;
#pragma unroll
      for (int m = 8; m; m >>= 1) rs += __shfl_xor(rs, m);
      lrow[i] = lrow[i] * al + rs;
      mrow[i] = mn;
      o0[i] *= al; o1[i] *= al; o2[i] *= al; o3[i] *= al;
      pw[(i + hf * 8) * 32 + col]      = (__bf16)p0;
      pw[(i + hf * 8) * 32 + 16 + col] = (__bf16)p1;
    }

    // O += P(16x32) . V(32xDH)
    const v16bf pa = load_frag(pw, 32, lane);
    o0 = wmma_bf16(pa, load_frag(vT_lds, 32, lane), o0);
    o1 = wmma_bf16(pa, load_frag(vT_lds + 16 * 32, 32, lane), o1);
    o2 = wmma_bf16(pa, load_frag(vT_lds + 32 * 32, 32, lane), o2);
    o3 = wmma_bf16(pa, load_frag(vT_lds + 48 * 32, 32, lane), o3);
  }

  // normalize + write out as [B,S,D] bf16
#pragma unroll
  for (int t = 0; t < 4; ++t) {
    v8f oc = (t == 0) ? o0 : (t == 1) ? o1 : (t == 2) ? o2 : o3;
#pragma unroll
    for (int i = 0; i < 8; ++i) {
      const int s = qbase + i + hf * 8;
      const int d = t * 16 + col;
      const float val = oc[i] / lrow[i];
      O[((size_t)(b * S_ + s)) * D_ + h * DH_ + d] = (__bf16)val;
    }
  }
}

// ---------------------------------------------------------------------------
// Host-side launch
// ---------------------------------------------------------------------------
extern "C" void kernel_launch(void* const* d_in, const int* in_sizes, int n_in,
                              void* d_out, int out_size, void* d_ws, size_t ws_size,
                              hipStream_t stream) {
  (void)in_sizes; (void)n_in; (void)out_size; (void)ws_size;

  const float* x     = (const float*)d_in[0];
  const float* ln1_g = (const float*)d_in[1];
  const float* ln1_b = (const float*)d_in[2];
  const float* Wq    = (const float*)d_in[3];
  const float* bq    = (const float*)d_in[4];
  const float* Wk    = (const float*)d_in[5];
  const float* bk    = (const float*)d_in[6];
  const float* Wv    = (const float*)d_in[7];
  const float* bv    = (const float*)d_in[8];
  const float* Wo    = (const float*)d_in[9];
  const float* bo    = (const float*)d_in[10];
  const float* ln2_g = (const float*)d_in[11];
  const float* ln2_b = (const float*)d_in[12];
  const float* W1    = (const float*)d_in[13];
  const float* b1    = (const float*)d_in[14];
  const float* W2    = (const float*)d_in[15];
  const float* b2    = (const float*)d_in[16];

  char* base = (char*)d_ws;
  size_t off = 0;
  auto alloc = [&](size_t bytes) -> void* {
    void* r = base + off;
    off += (bytes + 255) & ~(size_t)255;
    return r;
  };
  __bf16* Wq_t = (__bf16*)alloc((size_t)D_ * D_ * 2);   // [N][K]
  __bf16* Wk_t = (__bf16*)alloc((size_t)D_ * D_ * 2);
  __bf16* Wv_t = (__bf16*)alloc((size_t)D_ * D_ * 2);
  __bf16* Wo_t = (__bf16*)alloc((size_t)D_ * D_ * 2);
  __bf16* W1_t = (__bf16*)alloc((size_t)D_ * FF_ * 2);  // [FF][D]
  __bf16* W2_t = (__bf16*)alloc((size_t)FF_ * D_ * 2);  // [D][FF]
  __bf16* h1   = (__bf16*)alloc((size_t)ROWS_ * D_ * 2);
  __bf16* qb   = (__bf16*)alloc((size_t)ROWS_ * D_ * 2);  // [B,H,S,DH]
  __bf16* kb   = (__bf16*)alloc((size_t)ROWS_ * D_ * 2);  // [B,H,S,DH]
  __bf16* vb   = (__bf16*)alloc((size_t)ROWS_ * D_ * 2);  // [B,H,DH,S]
  __bf16* attn = (__bf16*)alloc((size_t)ROWS_ * D_ * 2);  // [B,S,D]
  float*  x1   = (float*) alloc((size_t)ROWS_ * D_ * 4);
  __bf16* h2   = (__bf16*)alloc((size_t)ROWS_ * D_ * 2);
  __bf16* gact = (__bf16*)alloc((size_t)ROWS_ * FF_ * 2);

  // 1) weights -> bf16, transposed to [N][K]
  auto cvt_t = [&](const float* src, __bf16* dst, int K, int N) {
    int n = K * N;
    cvt_bf16_t_kernel<<<(n + 255) / 256, 256, 0, stream>>>(src, dst, K, N);
  };
  cvt_t(Wq, Wq_t, D_, D_);
  cvt_t(Wk, Wk_t, D_, D_);
  cvt_t(Wv, Wv_t, D_, D_);
  cvt_t(Wo, Wo_t, D_, D_);
  cvt_t(W1, W1_t, D_, FF_);
  cvt_t(W2, W2_t, FF_, D_);

  // 2) LN1
  ln_bf16_kernel<<<ROWS_, 256, 0, stream>>>(x, ln1_g, ln1_b, h1);

  // 3) QKV projections (V scattered transposed)
  dim3 gD(D_ / 64, ROWS_ / 128);
  gemm_wmma_kernel<EPI_QKV><<<gD, 256, 0, stream>>>(h1, Wq_t, bq, nullptr, qb, ROWS_, D_, D_);
  gemm_wmma_kernel<EPI_QKV><<<gD, 256, 0, stream>>>(h1, Wk_t, bk, nullptr, kb, ROWS_, D_, D_);
  gemm_wmma_kernel<EPI_QKV_T><<<gD, 256, 0, stream>>>(h1, Wv_t, bv, nullptr, vb, ROWS_, D_, D_);

  // 4) causal flash attention
  attn_kernel<<<dim3(B_ * H_, S_ / 128), 256, 0, stream>>>(qb, kb, vb, attn);

  // 5) output projection + residual -> x1 (f32)
  gemm_wmma_kernel<EPI_RESID><<<gD, 256, 0, stream>>>(attn, Wo_t, bo, x, x1, ROWS_, D_, D_);

  // 6) LN2
  ln_bf16_kernel<<<ROWS_, 256, 0, stream>>>(x1, ln2_g, ln2_b, h2);

  // 7) FFN up + exact GELU -> bf16
  dim3 gF(FF_ / 64, ROWS_ / 128);
  gemm_wmma_kernel<EPI_GELU><<<gF, 256, 0, stream>>>(h2, W1_t, b1, nullptr, gact, ROWS_, FF_, D_);

  // 8) FFN down + bias + residual -> d_out (f32)
  gemm_wmma_kernel<EPI_RESID><<<gD, 256, 0, stream>>>(gact, W2_t, b2, x1, (float*)d_out,
                                                      ROWS_, D_, FF_);
}
